// MSDualBranchGPM_V2_27814208209679
// MI455X (gfx1250) — matrix operand order repaired and checked by reading, not compiled, not run on hardware
//
#include <hip/hip_runtime.h>
#include <hip/hip_bf16.h>
#include <climits>

typedef __attribute__((ext_vector_type(16))) _Float16 v16h;
typedef __attribute__((ext_vector_type(8)))  float    v8f;

union HF16 { v16h v; _Float16 h[16]; };
union FV8  { v8f  v; float f[8]; };
union U16x8 { uint4 u; _Float16 h[8]; };

#define NSEQ 2025
#define NPAD 2048
#define BSC 2
#define NHC 8
#define DAC 32
#define HIDC 1024
#define MROW (NSEQ*BSC)   // 4050
#define HW 45
#define MD 7
#define WSZ 15
#define NOFF 225

__device__ __forceinline__ v8f wmma_f16(v16h a, v16h b, v8f c) {
  return __builtin_amdgcn_wmma_f32_16x16x32_f16(false, a, false, b, (short)0, c, false, false);
}
__device__ __forceinline__ float silu_f(float x) { return x / (1.f + __expf(-x)); }

// ---------------------------------------------------------------------------
// Row LayerNorm over 256 cols, f16 output.  1 wave per row.
// ---------------------------------------------------------------------------
__global__ __launch_bounds__(32)
void ln_kernel(const float* __restrict__ X, const float* __restrict__ g,
               const float* __restrict__ bt, _Float16* __restrict__ Yh) {
  const int row = blockIdx.x, lane = threadIdx.x;
  const float* xr = X + (size_t)row * 256;
  float v[8], s = 0.f, sq = 0.f;
#pragma unroll
  for (int i = 0; i < 8; ++i) { int e = lane + 32 * i; v[i] = xr[e]; s += v[i]; sq += v[i] * v[i]; }
#pragma unroll
  for (int m = 1; m < 32; m <<= 1) { s += __shfl_xor(s, m, 32); sq += __shfl_xor(sq, m, 32); }
  float mean = s * (1.f / 256.f);
  float var  = sq * (1.f / 256.f) - mean * mean;
  float rs   = rsqrtf(var + 1e-5f);
#pragma unroll
  for (int i = 0; i < 8; ++i) {
    int e = lane + 32 * i;
    Yh[(size_t)row * 256 + e] = (_Float16)((v[i] - mean) * rs * g[e] + bt[e]);
  }
}

// ---------------------------------------------------------------------------
// f32 -> f16 cast
// ---------------------------------------------------------------------------
__global__ void cast_kernel(const float* __restrict__ in, _Float16* __restrict__ out, int n) {
  int i = blockIdx.x * blockDim.x + threadIdx.x;
  if (i < n) out[i] = (_Float16)in[i];
}

// ---------------------------------------------------------------------------
// WMMA GEMM: C[M,Nc] = act(A[M,K]f16 @ B[K,Nc]f16 + bias), silu on cols>=split
// block: 256 threads = 8 waves; block tile 64x64; each wave: one A-frag,
// two 16x16 N-tiles -> 2 WMMA per K-step.  Vectorized b128 staging.
// ---------------------------------------------------------------------------
#define BM 64
#define BN 64
#define BK 32
#define LDA 40   // halfs; 80B row stride (16B aligned)
#define LDB 40
__global__ __launch_bounds__(256)
void gemm_kernel(const _Float16* __restrict__ A, const _Float16* __restrict__ B,
                 const float* __restrict__ bias, float* __restrict__ C,
                 int M, int K, int Nc, int act_split) {
  __shared__ _Float16 sA[BM][LDA];   // row-major tile [m][k]
  __shared__ _Float16 sB[BN][LDB];   // transposed: [n][k]
  const int tid = threadIdx.x;
  const int lane = tid & 31, wid = tid >> 5;
  const int wR = wid & 3, wC = wid >> 2;     // wR: 4 row tiles, wC: 2 col pairs
  const int col = lane & 15, hi = lane >> 4;
  const int kb0 = hi ? 16 : 0, kb1 = hi ? 24 : 8, kbB = hi ? 16 : 0;
  const int m0 = blockIdx.x * BM, n0 = blockIdx.y * BN;
  const int arow = tid >> 2, aseg = (tid & 3) * 8;   // A: 64 rows x 4 segs of 8 halfs
  const int brow = tid >> 3, bseg = (tid & 7) * 8;   // B: 32 rows x 8 segs of 8 halfs

  FV8 acc0, acc1;
#pragma unroll
  for (int r = 0; r < 8; ++r) { acc0.f[r] = 0.f; acc1.f[r] = 0.f; }

  for (int k0 = 0; k0 < K; k0 += BK) {
    { // stage A: one b128 global load + one b128 LDS store per thread
      int gm = m0 + arow;
      U16x8 val;
      val.u.x = 0u; val.u.y = 0u; val.u.z = 0u; val.u.w = 0u;
      const _Float16* ap = A + (size_t)gm * K + k0 + aseg;
      if (gm < M) {
        val.u = *(const uint4*)ap;
        if (k0 + BK < K) __builtin_prefetch(ap + BK, 0, 3);
      }
      *(uint4*)&sA[arow][aseg] = val.u;
    }
    { // stage B: b128 global load, transpose-scatter into [n][k]
      const _Float16* bp = B + (size_t)(k0 + brow) * Nc + n0 + bseg;
      U16x8 val;
      val.u = *(const uint4*)bp;
      if (k0 + BK < K) __builtin_prefetch(bp + (size_t)BK * Nc, 0, 3);
#pragma unroll
      for (int i = 0; i < 8; ++i) sB[bseg + i][brow] = val.h[i];
    }
    __syncthreads();
    HF16 a, bf0, bf1;
    const _Float16* arp = &sA[wR * 16 + col][0];
#pragma unroll
    for (int e = 0; e < 8; ++e) { a.h[e] = arp[kb0 + e]; a.h[8 + e] = arp[kb1 + e]; }
    const _Float16* brp0 = &sB[wC * 32 + col][kbB];
    const _Float16* brp1 = &sB[wC * 32 + 16 + col][kbB];
#pragma unroll
    for (int e = 0; e < 16; ++e) { bf0.h[e] = brp0[e]; bf1.h[e] = brp1[e]; }
    acc0.v = wmma_f16(a.v, bf0.v, acc0.v);
    acc1.v = wmma_f16(a.v, bf1.v, acc1.v);
    __syncthreads();
  }

  const int cn0 = n0 + wC * 32 + col;
  const int cn1 = cn0 + 16;
  const float bs0 = bias ? bias[cn0] : 0.f;
  const float bs1 = bias ? bias[cn1] : 0.f;
#pragma unroll
  for (int r = 0; r < 8; ++r) {
    int cm = m0 + wR * 16 + r + (hi << 3);
    if (cm < M) {
      float v0 = acc0.f[r] + bs0;
      float v1 = acc1.f[r] + bs1;
      if (cn0 >= act_split) v0 = silu_f(v0);
      if (cn1 >= act_split) v1 = silu_f(v1);
      C[(size_t)cm * Nc + cn0] = v0;
      C[(size_t)cm * Nc + cn1] = v1;
    }
  }
}

// ---------------------------------------------------------------------------
// Pack Q/K (f16, [bh][NPAD][32]) and V (transposed [bh][128][NPAD], optional
// row-major [bh][NPAD][128]) from f32 activation matrices.
// ---------------------------------------------------------------------------
__global__ __launch_bounds__(128)
void pack_kernel(const float* __restrict__ Qsrc, int ldq,
                 const float* __restrict__ VA, int ldva, int va_off, int splitc,
                 const float* __restrict__ VB, int ldvb, int vb_off,
                 _Float16* __restrict__ Qh, _Float16* __restrict__ Kh,
                 _Float16* __restrict__ Vt, _Float16* __restrict__ Vh) {
  const int e = threadIdx.x;
  const int n = blockIdx.x;        // 0..NPAD-1
  const int bh = blockIdx.y;
  const int b = bh >> 3, h = bh & 7;
  const size_t row = (size_t)n * BSC + b;
  const int c = h * 128 + e;
  float vv = 0.f;
  if (n < NSEQ)
    vv = (c < splitc) ? VA[row * ldva + va_off + c] : VB[row * ldvb + (c + vb_off)];
  Vt[((size_t)bh * 128 + e) * NPAD + n] = (_Float16)vv;
  if (Vh) Vh[((size_t)bh * NPAD + n) * 128 + e] = (_Float16)vv;
  if (e < 32) {
    float qf = (n < NSEQ) ? Qsrc[row * ldq + h * 32 + e] : 0.f;
    Qh[((size_t)bh * NPAD + n) * 32 + e] = (_Float16)(qf * 0.17677669529663687f); // 1/sqrt(32)
    Kh[((size_t)bh * NPAD + n) * 32 + e] = (_Float16)qf;
  }
}

// ---------------------------------------------------------------------------
// Global attention (flash-style), one wave per (16 query rows, bh).
// Qh scaled f16, Kh f16, Vt transposed f16 padded to NPAD.  Out f32 [M][1024].
// ---------------------------------------------------------------------------
__global__ __launch_bounds__(32)
void attn_global_kernel(const _Float16* __restrict__ Qh, const _Float16* __restrict__ Kh,
                        const _Float16* __restrict__ Vt, float* __restrict__ Out) {
  __shared__ _Float16 pbuf[16 * 32];
  const int lane = threadIdx.x;
  const int tile = blockIdx.x;        // 0..126
  const int bh = blockIdx.y;          // 0..15
  const int b = bh >> 3, h = bh & 7;
  const int col = lane & 15, hi = lane >> 4;
  const int kb0 = hi ? 16 : 0, kb1 = hi ? 24 : 8, kbB = hi ? 16 : 0;
  const v8f vzero = {0.f, 0.f, 0.f, 0.f, 0.f, 0.f, 0.f, 0.f};

  HF16 aq;
  {
    const _Float16* qrow = Qh + ((size_t)bh * NPAD + (tile * 16 + col)) * 32;
#pragma unroll
    for (int e = 0; e < 8; ++e) { aq.h[e] = qrow[kb0 + e]; aq.h[8 + e] = qrow[kb1 + e]; }
  }

  FV8 acc[8];
#pragma unroll
  for (int t = 0; t < 8; ++t)
#pragma unroll
    for (int r = 0; r < 8; ++r) acc[t].f[r] = 0.f;
  float mrow[8], lrow[8];
#pragma unroll
  for (int r = 0; r < 8; ++r) { mrow[r] = -1e30f; lrow[r] = 0.f; }

  for (int mb = 0; mb < NPAD; mb += 32) {
    FV8 s0, s1;
    {
      const _Float16* kp0 = Kh + ((size_t)bh * NPAD + (mb + col)) * 32 + kbB;
      const _Float16* kp1 = Kh + ((size_t)bh * NPAD + (mb + 16 + col)) * 32 + kbB;
      v16h bk0 = *(const v16h*)kp0;
      v16h bk1 = *(const v16h*)kp1;
      s0.v = wmma_f16(aq.v, bk0, vzero);
      s1.v = wmma_f16(aq.v, bk1, vzero);
    }
    if (mb + col >= NSEQ) {
#pragma unroll
      for (int r = 0; r < 8; ++r) s0.f[r] = -1e9f;
    }
    if (mb + 16 + col >= NSEQ) {
#pragma unroll
      for (int r = 0; r < 8; ++r) s1.f[r] = -1e9f;
    }

    float p0[8], p1[8], sc[8];
#pragma unroll
    for (int r = 0; r < 8; ++r) {
      float v = fmaxf(s0.f[r], s1.f[r]);
#pragma unroll
      for (int m = 1; m < 16; m <<= 1) v = fmaxf(v, __shfl_xor(v, m, 32));
      float mn = fmaxf(mrow[r], v);
      sc[r] = __expf(mrow[r] - mn);
      mrow[r] = mn;
      p0[r] = __expf(s0.f[r] - mn);
      p1[r] = __expf(s1.f[r] - mn);
      float rs = p0[r] + p1[r];
#pragma unroll
      for (int m = 1; m < 16; m <<= 1) rs += __shfl_xor(rs, m, 32);
      lrow[r] = lrow[r] * sc[r] + rs;
    }
#pragma unroll
    for (int t = 0; t < 8; ++t)
#pragma unroll
      for (int r = 0; r < 8; ++r) acc[t].f[r] *= sc[r];

    __syncthreads();  // protect pbuf from previous iteration readers
#pragma unroll
    for (int r = 0; r < 8; ++r) {
      int m = r + (hi << 3);
      pbuf[m * 32 + col]      = (_Float16)p0[r];
      pbuf[m * 32 + 16 + col] = (_Float16)p1[r];
    }
    __syncthreads();
    HF16 ap;
#pragma unroll
    for (int e = 0; e < 8; ++e) {
      ap.h[e]     = pbuf[col * 32 + kb0 + e];
      ap.h[8 + e] = pbuf[col * 32 + kb1 + e];
    }
#pragma unroll
    for (int t = 0; t < 8; ++t) {
      const _Float16* vp = Vt + ((size_t)bh * 128 + t * 16 + col) * NPAD + mb + kbB;
      v16h bv = *(const v16h*)vp;
      acc[t].v = wmma_f16(ap.v, bv, acc[t].v);
    }
  }

  float inv[8];
#pragma unroll
  for (int r = 0; r < 8; ++r) inv[r] = 1.f / lrow[r];
#pragma unroll
  for (int r = 0; r < 8; ++r) {
    int m = tile * 16 + r + (hi << 3);
    if (m < NSEQ) {
      float* orow = Out + ((size_t)m * BSC + b) * HIDC + h * 128;
#pragma unroll
      for (int t = 0; t < 8; ++t) orow[t * 16 + col] = acc[t].f[r] * inv[r];
    }
  }
}

// ---------------------------------------------------------------------------
// Local 15x15 windowed attention, one wave per (pixel, bh).
// score = q.(k + Wrel_row) + brel; softmax over 225 offsets; agg 128-d V.
// ---------------------------------------------------------------------------
__global__ __launch_bounds__(32)
void attn_local_kernel(const _Float16* __restrict__ Qh, const _Float16* __restrict__ Kh,
                       const _Float16* __restrict__ Vh, const float* __restrict__ Wrel,
                       const float* __restrict__ brel, float* __restrict__ Out) {
  __shared__ float sbuf[256];
  const int lane = threadIdx.x;
  const int n = blockIdx.x;           // pixel
  const int bh = blockIdx.y;
  const int b = bh >> 3, h = bh & 7;
  const int y = n / HW, x = n % HW;

  float q[32];
  const _Float16* qrow = Qh + ((size_t)bh * NPAD + n) * 32;
#pragma unroll
  for (int d = 0; d < 32; ++d) q[d] = (float)qrow[d];

  float sreg[8];
#pragma unroll
  for (int oi = 0; oi < 8; ++oi) {
    int o = lane + 32 * oi;
    float s = -1e9f;
    if (o < NOFF) {
      int dy = o / WSZ - MD, dx = o % WSZ - MD;
      int yy = y + dy, xx = x + dx;
      if (yy >= 0 && yy < HW && xx >= 0 && xx < HW) {
        int n2 = yy * HW + xx;
        const _Float16* kr = Kh + ((size_t)bh * NPAD + n2) * 32;
        const float* wr = Wrel + ((size_t)h * NOFF + o) * 32;
        float a = brel[h * NOFF + o];
#pragma unroll
        for (int d = 0; d < 32; ++d) a += q[d] * ((float)kr[d] + wr[d]);
        s = a;
      }
    }
    sreg[oi] = s;
  }
  float mx = sreg[0];
#pragma unroll
  for (int oi = 1; oi < 8; ++oi) mx = fmaxf(mx, sreg[oi]);
#pragma unroll
  for (int m = 1; m < 32; m <<= 1) mx = fmaxf(mx, __shfl_xor(mx, m, 32));
  float p[8], sum = 0.f;
#pragma unroll
  for (int oi = 0; oi < 8; ++oi) { p[oi] = __expf(sreg[oi] - mx); sum += p[oi]; }
#pragma unroll
  for (int m = 1; m < 32; m <<= 1) sum += __shfl_xor(sum, m, 32);
  float inv = 1.f / sum;
#pragma unroll
  for (int oi = 0; oi < 8; ++oi) sbuf[lane + 32 * oi] = p[oi] * inv;
  __syncthreads();

  const int e0 = lane * 4;
  float acc4[4] = {0.f, 0.f, 0.f, 0.f};
  for (int o = 0; o < NOFF; ++o) {
    float a = sbuf[o];
    if (a != 0.f) {
      int dy = o / WSZ - MD, dx = o % WSZ - MD;
      int n2 = (y + dy) * HW + (x + dx);
      const _Float16* vr = Vh + ((size_t)bh * NPAD + n2) * 128 + e0;
#pragma unroll
      for (int i = 0; i < 4; ++i) acc4[i] += a * (float)vr[i];
    }
  }
  float* orow = Out + ((size_t)n * BSC + b) * HIDC + h * 128 + e0;
#pragma unroll
  for (int i = 0; i < 4; ++i) orow[i] = acc4[i];
}

// ---------------------------------------------------------------------------
// Gating: mode 1: gate = e<512 ? silu(U512[row,e]) : 1   (cat_U)
//         mode 2: gate = U1024[row,e]                    (already silu'd Us)
// ---------------------------------------------------------------------------
__global__ void gate_kernel(const float* __restrict__ attn, const float* __restrict__ U,
                            _Float16* __restrict__ out, int mode, int total) {
  int idx = blockIdx.x * blockDim.x + threadIdx.x;
  if (idx >= total) return;
  int row = idx >> 10, e = idx & 1023;
  float g;
  if (mode == 1) g = (e < 512) ? silu_f(U[(size_t)row * 512 + e]) : 1.f;
  else           g = U[idx];
  out[idx] = (_Float16)(attn[idx] * g);
}

// ---------------------------------------------------------------------------
// Combine residuals + two LayerNorms -> x (f16 [M][512]); keep f32 residuals.
// ---------------------------------------------------------------------------
__global__ __launch_bounds__(32)
void combine_kernel(const float* __restrict__ tgt, const float* __restrict__ t2,
                    const float* __restrict__ t3, const float* __restrict__ g2,
                    const float* __restrict__ b2, const float* __restrict__ gi,
                    const float* __restrict__ bi, float* __restrict__ ntgt,
                    float* __restrict__ tid_, _Float16* __restrict__ xh) {
  const int row = blockIdx.x, lane = threadIdx.x;
  float nv[8], iv[8];
  float s1 = 0.f, q1 = 0.f, s2 = 0.f, q2 = 0.f;
#pragma unroll
  for (int i = 0; i < 8; ++i) {
    int e = lane + 32 * i;
    float a = t2[(size_t)row * 512 + e] + t3[(size_t)row * 512 + e];
    float c = t2[(size_t)row * 512 + 256 + e] + t3[(size_t)row * 512 + 256 + e];
    nv[i] = tgt[(size_t)row * 256 + e] + a;
    iv[i] = c;
    s1 += nv[i]; q1 += nv[i] * nv[i];
    s2 += iv[i]; q2 += iv[i] * iv[i];
  }
#pragma unroll
  for (int m = 1; m < 32; m <<= 1) {
    s1 += __shfl_xor(s1, m, 32); q1 += __shfl_xor(q1, m, 32);
    s2 += __shfl_xor(s2, m, 32); q2 += __shfl_xor(q2, m, 32);
  }
  float m1 = s1 * (1.f / 256.f), v1 = q1 * (1.f / 256.f) - m1 * m1, r1 = rsqrtf(v1 + 1e-5f);
  float m2 = s2 * (1.f / 256.f), v2 = q2 * (1.f / 256.f) - m2 * m2, r2 = rsqrtf(v2 + 1e-5f);
#pragma unroll
  for (int i = 0; i < 8; ++i) {
    int e = lane + 32 * i;
    ntgt[(size_t)row * 256 + e] = nv[i];
    tid_[(size_t)row * 256 + e] = iv[i];
    xh[(size_t)row * 512 + e]       = (_Float16)((nv[i] - m1) * r1 * g2[e] + b2[e]);
    xh[(size_t)row * 512 + 256 + e] = (_Float16)((iv[i] - m2) * r2 * gi[e] + bi[e]);
  }
}

// ---------------------------------------------------------------------------
// Final residual add, write concatenated outputs.
// ---------------------------------------------------------------------------
__global__ void final_kernel(const float* __restrict__ ntgt, const float* __restrict__ tid_,
                             const float* __restrict__ s2, float* __restrict__ out, int total) {
  int idx = blockIdx.x * blockDim.x + threadIdx.x;
  if (idx >= total) return;
  int row = idx >> 8, e = idx & 255;
  out[idx]         = ntgt[idx] + s2[(size_t)row * 512 + e];
  out[total + idx] = tid_[idx] + s2[(size_t)row * 512 + 256 + e];
}

// ---------------------------------------------------------------------------
extern "C" void kernel_launch(void* const* d_in, const int* in_sizes, int n_in,
                              void* d_out, int out_size, void* d_ws, size_t ws_size,
                              hipStream_t stream) {
  (void)in_sizes; (void)n_in; (void)out_size; (void)ws_size;
  const float* tgt    = (const float*)d_in[0];
  const float* idemb  = (const float*)d_in[1];
  const float* ln1g   = (const float*)d_in[2];
  const float* ln1b   = (const float*)d_in[3];
  const float* Wqv    = (const float*)d_in[4];
  const float* bqv    = (const float*)d_in[5];
  const float* Wu     = (const float*)d_in[6];
  const float* bu     = (const float*)d_in[7];
  const float* Widv   = (const float*)d_in[8];
  const float* bidv   = (const float*)d_in[9];
  const float* ltWo   = (const float*)d_in[10];
  const float* ltbo   = (const float*)d_in[11];
  const float* Wrel   = (const float*)d_in[12];
  const float* brel   = (const float*)d_in[13];
  const float* stWo   = (const float*)d_in[14];
  const float* stbo   = (const float*)d_in[15];
  const float* ln2g   = (const float*)d_in[16];
  const float* ln2b   = (const float*)d_in[17];
  const float* idg    = (const float*)d_in[18];
  const float* idb    = (const float*)d_in[19];
  const float* saWqk  = (const float*)d_in[20];
  const float* sabqk  = (const float*)d_in[21];
  const float* saWv   = (const float*)d_in[22];
  const float* sabv   = (const float*)d_in[23];
  const float* saWu   = (const float*)d_in[24];
  const float* sabu   = (const float*)d_in[25];
  const float* saWo   = (const float*)d_in[26];
  const float* sabo   = (const float*)d_in[27];
  float* outp = (float*)d_out;

  char* wsb = (char*)d_ws;
  size_t off = 0;
  auto alloc = [&](size_t bytes) -> void* {
    off = (off + 255) & ~(size_t)255;
    void* p = wsb + off;
    off += bytes;
    return p;
  };
  const size_t M = MROW;
  _Float16* xln_h   = (_Float16*)alloc(M * 256 * 2);
  _Float16* idemb_h = (_Float16*)alloc(M * 256 * 2);
  _Float16* wqv_h   = (_Float16*)alloc(256 * 768 * 2);
  _Float16* wu_h    = (_Float16*)alloc(256 * 512 * 2);
  _Float16* widv_h  = (_Float16*)alloc(256 * 512 * 2);
  _Float16* ltwo_h  = (_Float16*)alloc(1024 * 512 * 2);
  _Float16* stwo_h  = (_Float16*)alloc(1024 * 512 * 2);
  _Float16* saqk_h  = (_Float16*)alloc(512 * 256 * 2);
  _Float16* savv_h  = (_Float16*)alloc(512 * 1024 * 2);
  _Float16* sauu_h  = (_Float16*)alloc(512 * 1024 * 2);
  _Float16* sawo_h  = (_Float16*)alloc(1024 * 512 * 2);
  float*    qv      = (float*)alloc(M * 768 * 4);
  float*    cu      = (float*)alloc(M * 512 * 4);
  float*    gidv    = (float*)alloc(M * 512 * 4);
  _Float16* Qhb     = (_Float16*)alloc((size_t)16 * NPAD * 32 * 2);
  _Float16* Khb     = (_Float16*)alloc((size_t)16 * NPAD * 32 * 2);
  _Float16* Vtb     = (_Float16*)alloc((size_t)16 * 128 * NPAD * 2);
  _Float16* Vhb     = (_Float16*)alloc((size_t)16 * NPAD * 128 * 2);
  float*    attn    = (float*)alloc(M * 1024 * 4);   // reused for attention #2
  float*    attnL   = (float*)alloc(M * 1024 * 4);
  _Float16* gated   = (_Float16*)alloc(M * 1024 * 2); // reused
  _Float16* gatedL  = (_Float16*)alloc(M * 1024 * 2);
  float*    t2      = (float*)alloc(M * 512 * 4);    // reused for s2
  float*    t3      = (float*)alloc(M * 512 * 4);
  float*    ntgt    = (float*)alloc(M * 256 * 4);
  float*    tid_    = (float*)alloc(M * 256 * 4);
  _Float16* xh      = (_Float16*)alloc(M * 512 * 2);
  float*    Qs      = (float*)alloc(M * 256 * 4);
  float*    Vs      = (float*)alloc(M * 1024 * 4);
  float*    Us      = (float*)alloc(M * 1024 * 4);

  auto cast = [&](const float* src, _Float16* dst, int n) {
    cast_kernel<<<(n + 255) / 256, 256, 0, stream>>>(src, dst, n);
  };
  auto gemm = [&](const _Float16* A, const _Float16* B, const float* bias, float* C,
                  int m, int k, int nc, int split) {
    dim3 g((m + BM - 1) / BM, nc / BN);
    gemm_kernel<<<g, 256, 0, stream>>>(A, B, bias, C, m, k, nc, split);
  };

  // Stage 0: LN + f16 casts
  ln_kernel<<<M, 32, 0, stream>>>(tgt, ln1g, ln1b, xln_h);
  cast(idemb, idemb_h, (int)(M * 256));
  cast(Wqv,  wqv_h,  256 * 768);
  cast(Wu,   wu_h,   256 * 512);
  cast(Widv, widv_h, 256 * 512);
  cast(ltWo, ltwo_h, 1024 * 512);
  cast(stWo, stwo_h, 1024 * 512);
  cast(saWqk, saqk_h, 512 * 256);
  cast(saWv,  savv_h, 512 * 1024);
  cast(saWu,  sauu_h, 512 * 1024);
  cast(saWo,  sawo_h, 1024 * 512);

  // Stage 1: projections
  gemm(xln_h,   wqv_h,  bqv,  qv,   (int)M, 256, 768, 256);      // silu on cols>=256 (curr_V)
  gemm(xln_h,   wu_h,   bu,   cu,   (int)M, 256, 512, INT_MAX);  // curr_U (no act)
  gemm(idemb_h, widv_h, bidv, gidv, (int)M, 256, 512, 0);        // g_ID_V = silu(...)

  // Stage 2: pack Q/K/V and run global + local attention
  {
    dim3 g(NPAD, 16);
    pack_kernel<<<g, 128, 0, stream>>>(qv, 768, qv, 768, 256, 512, gidv, 512, -512,
                                       Qhb, Khb, Vtb, Vhb);
  }
  {
    dim3 g((NSEQ + 15) / 16, 16);
    attn_global_kernel<<<g, 32, 0, stream>>>(Qhb, Khb, Vtb, attn);
  }
  {
    dim3 g(NSEQ, 16);
    attn_local_kernel<<<g, 32, 0, stream>>>(Qhb, Khb, Vhb, Wrel, brel, attnL);
  }

  // Stage 3: gating + output projections
  gate_kernel<<<((int)(M * 1024) + 255) / 256, 256, 0, stream>>>(attn, cu, gated, 1, (int)(M * 1024));
  gate_kernel<<<((int)(M * 1024) + 255) / 256, 256, 0, stream>>>(attnL, cu, gatedL, 1, (int)(M * 1024));
  gemm(gated,  ltwo_h, ltbo, t2, (int)M, 1024, 512, INT_MAX);
  gemm(gatedL, stwo_h, stbo, t3, (int)M, 1024, 512, INT_MAX);

  // Stage 4: residual combine + two LNs -> x
  combine_kernel<<<M, 32, 0, stream>>>(tgt, t2, t3, ln2g, ln2b, idg, idb, ntgt, tid_, xh);

  // Stage 5: second attention block
  gemm(xh, saqk_h, sabqk, Qs, (int)M, 512, 256, INT_MAX);
  gemm(xh, savv_h, sabv,  Vs, (int)M, 512, 1024, 0);
  gemm(xh, sauu_h, sabu,  Us, (int)M, 512, 1024, 0);
  {
    dim3 g(NPAD, 16);
    pack_kernel<<<g, 128, 0, stream>>>(Qs, 256, Vs, 1024, 0, 1024, Vs, 1024, 0,
                                       Qhb, Khb, Vtb, (_Float16*)nullptr);
  }
  {
    dim3 g((NSEQ + 15) / 16, 16);
    attn_global_kernel<<<g, 32, 0, stream>>>(Qhb, Khb, Vtb, attn);
  }
  gate_kernel<<<((int)(M * 1024) + 255) / 256, 256, 0, stream>>>(attn, Us, gated, 2, (int)(M * 1024));
  gemm(gated, sawo_h, sabo, t2, (int)M, 1024, 512, INT_MAX);  // s2 into t2 buffer

  // Stage 6: final residuals
  final_kernel<<<((int)(M * 256) + 255) / 256, 256, 0, stream>>>(ntgt, tid_, t2, outp, (int)(M * 256));
}
